// IRMVSNet_79517024518234
// MI455X (gfx1250) — matrix operand order repaired and checked by reading, not compiled, not run on hardware
//
#include <hip/hip_runtime.h>
#include <hip/hip_bf16.h>

typedef __attribute__((ext_vector_type(16))) _Float16 v16h;
typedef __attribute__((ext_vector_type(8)))  _Float16 v8h;
typedef __attribute__((ext_vector_type(8)))  float    v8f;

#define IH 128
#define IW 160
#define IHW (IH * IW)
#define NDEPTH 96

// ---------------------------------------------------------------------------
// 0. Hw[d][j] = inv(Hs[j, M-1]) @ Hs[j, d]     (Hs: (2, 96, 3, 3))
// ---------------------------------------------------------------------------
__global__ void compute_hw_kernel(const float* __restrict__ Hs,
                                  float* __restrict__ Hw, int M) {
  __shared__ float inv[2][9];
  int t = threadIdx.x;
  if (t < 2) {
    const float* A = Hs + (size_t)(t * M + (M - 1)) * 9;
    float a = A[0], b = A[1], c = A[2], d = A[3], e = A[4], f = A[5],
          g = A[6], h = A[7], i = A[8];
    float A11 = e * i - f * h;
    float A12 = f * g - d * i;
    float A13 = d * h - e * g;
    float det = a * A11 + b * A12 + c * A13;
    float id = 1.0f / det;
    inv[t][0] = A11 * id; inv[t][1] = (c * h - b * i) * id; inv[t][2] = (b * f - c * e) * id;
    inv[t][3] = A12 * id; inv[t][4] = (a * i - c * g) * id; inv[t][5] = (c * d - a * f) * id;
    inv[t][6] = A13 * id; inv[t][7] = (b * g - a * h) * id; inv[t][8] = (a * e - b * d) * id;
  }
  __syncthreads();
  if (t < 2 * M) {
    int dpt = t >> 1, j = t & 1;
    const float* B = Hs + (size_t)(j * M + dpt) * 9;
    float* O = Hw + (size_t)(dpt * 2 + j) * 9;
    #pragma unroll
    for (int r = 0; r < 3; ++r)
      #pragma unroll
      for (int cc = 0; cc < 3; ++cc) {
        float s = 0.f;
        #pragma unroll
        for (int kk = 0; kk < 3; ++kk) s += inv[j][r * 3 + kk] * B[kk * 3 + cc];
        O[r * 3 + cc] = s;
      }
  }
}

// ---------------------------------------------------------------------------
// 1. Zero-init
// ---------------------------------------------------------------------------
__global__ void zero_kernel(float* __restrict__ p, int n) {
  int i = blockIdx.x * 256 + threadIdx.x;
  if (i < n) p[i] = 0.0f;
}

// ---------------------------------------------------------------------------
// 2. Pack weights (Cout, Ctot, 3, 3) f32 -> f16 [Npad][Kpad], tap-major:
//    k = tap*CP + c  (CP = padded channel count), matches A-fragment groups.
// ---------------------------------------------------------------------------
__global__ void pack_w_kernel(const float* __restrict__ w, _Float16* __restrict__ dst,
                              int Cout, int Ctot, int CP, int Kpad, int Npad) {
  int i = blockIdx.x * 256 + threadIdx.x;
  int total = Npad * Kpad;
  if (i >= total) return;
  int n = i / Kpad;
  int k = i - n * Kpad;
  int t = k / CP;
  int c = k - t * CP;
  float v = (n < Cout && t < 9 && c < Ctot)
                ? w[((size_t)n * Ctot + c) * 9 + t] : 0.0f;
  dst[i] = (_Float16)v;
}

// ---------------------------------------------------------------------------
// 3. Warp + variance cost; x = -cost -> f16 into both GRU1 conv inputs
// ---------------------------------------------------------------------------
__global__ __launch_bounds__(256) void warp_cost_kernel(
    const float* __restrict__ feat,   // (3, 32, H, W) f32
    const float* __restrict__ HwAll,  // (96, 2, 9)
    int d,
    _Float16* __restrict__ xg,        // [IHW][48]
    _Float16* __restrict__ xo) {      // [IHW][48]
  int p = blockIdx.x * 256 + threadIdx.x;
  if (p >= IHW) return;
  float x = (float)(p % IW);
  float y = (float)(p / IW);
  float s1[32], s2[32];
  #pragma unroll
  for (int c = 0; c < 32; ++c) {
    float v = feat[(size_t)c * IHW + p];
    s1[c] = v; s2[c] = v * v;
  }
  for (int j = 0; j < 2; ++j) {
    const float* Hm = HwAll + (size_t)(d * 2 + j) * 9;
    float w  = Hm[6] * x + Hm[7] * y + Hm[8];
    float iw = 1.0f / w;
    float u  = (Hm[0] * x + Hm[1] * y + Hm[2]) * iw;
    float v  = (Hm[3] * x + Hm[4] * y + Hm[5]) * iw;
    float u0 = floorf(u), v0 = floorf(v);
    float du = u - u0, dv = v - v0;
    float wts[4] = {(1.f - du) * (1.f - dv), du * (1.f - dv),
                    (1.f - du) * dv,          du * dv};
    float us[4] = {u0, u0 + 1.f, u0, u0 + 1.f};
    float vs[4] = {v0, v0, v0 + 1.f, v0 + 1.f};
    int   idxs[4];
    float wv[4];
    #pragma unroll
    for (int t = 0; t < 4; ++t) {
      float ui = us[t], vi = vs[t];
      float valid = (ui >= 0.f && ui <= (float)(IW - 1) &&
                     vi >= 0.f && vi <= (float)(IH - 1)) ? 1.0f : 0.0f;
      int uic = (int)fminf(fmaxf(ui, 0.f), (float)(IW - 1));
      int vic = (int)fminf(fmaxf(vi, 0.f), (float)(IH - 1));
      idxs[t] = vic * IW + uic;
      wv[t]   = valid * wts[t];
    }
    const float* fj = feat + (size_t)(j + 1) * 32 * IHW;
    #pragma unroll
    for (int c = 0; c < 32; ++c) {
      const float* fc = fj + (size_t)c * IHW;
      float val = fc[idxs[0]] * wv[0] + fc[idxs[1]] * wv[1] +
                  fc[idxs[2]] * wv[2] + fc[idxs[3]] * wv[3];
      s1[c] += val; s2[c] += val * val;
    }
  }
  const float inv3 = 1.0f / 3.0f;
  _Float16* xgp = xg + (size_t)p * 48;
  _Float16* xop = xo + (size_t)p * 48;
  #pragma unroll
  for (int c = 0; c < 32; ++c) {
    float m = s1[c] * inv3;
    _Float16 hv = (_Float16)(-(s2[c] * inv3 - m * m));
    xgp[c] = hv;
    xop[c] = hv;
  }
}

// ---------------------------------------------------------------------------
// 4. Implicit-GEMM 3x3 SAME conv, v_wmma_f32_16x16x32_f16.
//    Channel-last f16 input [IHW][CP], zero-padded channels.
//    Wave = MT m-tiles (16 px each) x NT n-tiles (16 ch each).
//    K loop fully unrolled (KC compile-time) -> tap/channel constant-folded,
//    branchless clamped loads + cndmask zeroing (no EXEC manipulation).
// ---------------------------------------------------------------------------
template <int NT, int MT, int CP>
__global__ __launch_bounds__(128) void conv3x3_wmma_kernel(
    const _Float16* __restrict__ in,   // [IHW][CP] f16
    const _Float16* __restrict__ Wpk,  // [Npad][KP] f16
    const float* __restrict__ bias,
    float* __restrict__ out, int Cout) {
  constexpr int KP = ((9 * CP + 31) / 32) * 32;
  constexpr int KC = KP / 32;

  const int lane   = threadIdx.x & 31;
  const int wid    = threadIdx.x >> 5;
  const int mBase0 = (blockIdx.x * 4 + wid) * (16 * MT);
  if (mBase0 >= IHW) return;  // uniform per wave

  const int laneHi = (lane >> 4) & 1;
  const int nLane  = lane & 15;

  int py[MT], px[MT];
  #pragma unroll
  for (int mt = 0; mt < MT; ++mt) {
    int m = mBase0 + mt * 16 + nLane;
    py[mt] = m / IW;
    px[mt] = m - py[mt] * IW;
  }

  v8f acc[MT][NT] = {};

  #pragma unroll
  for (int kc = 0; kc < KC; ++kc) {
    // --- B fragments (shared across m-tiles) ---
    v16h b[NT];
    #pragma unroll
    for (int nt = 0; nt < NT; ++nt) {
      const _Float16* bp = Wpk + (size_t)((nt << 4) + nLane) * KP +
                           (kc << 5) + (laneHi << 3);
      v8h b0 = *(const v8h*)bp;
      v8h b1 = *(const v8h*)(bp + 16);
      b[nt] = __builtin_shufflevector(b0, b1, 0, 1, 2, 3, 4, 5, 6, 7,
                                      8, 9, 10, 11, 12, 13, 14, 15);  // two 16B-aligned 8-k groups
    }
    #pragma unroll
    for (int mt = 0; mt < MT; ++mt) {
      // --- A fragment: two 8-channel groups, branchless ---
      v8h g[2];
      #pragma unroll
      for (int gi = 0; gi < 2; ++gi) {
        const int kg = kc * 32 + gi * 16 + laneHi * 8;  // compile-time const
        const int t  = kg / CP;
        const int c  = kg - t * CP;
        v8h z = {};
        if (t >= 9) {
          g[gi] = z;
        } else {
          const int dy = t / 3 - 1;
          const int dx = t - (t / 3) * 3 - 1;
          int yy = py[mt] + dy, xx = px[mt] + dx;
          bool ok = ((unsigned)yy < (unsigned)IH) & ((unsigned)xx < (unsigned)IW);
          int yc = yy < 0 ? 0 : (yy > IH - 1 ? IH - 1 : yy);
          int xc = xx < 0 ? 0 : (xx > IW - 1 ? IW - 1 : xx);
          v8h v = *(const v8h*)(in + (size_t)(yc * IW + xc) * CP + c);
          g[gi] = ok ? v : z;
        }
      }
      v16h a = __builtin_shufflevector(g[0], g[1], 0, 1, 2, 3, 4, 5, 6, 7,
                                       8, 9, 10, 11, 12, 13, 14, 15);
      #pragma unroll
      for (int nt = 0; nt < NT; ++nt)
        acc[mt][nt] = __builtin_amdgcn_wmma_f32_16x16x32_f16(
            false, a, false, b[nt], (short)0, acc[mt][nt], false, false);
    }
  }

  // C layout: VGPR r -> M = r + 8*laneHi, N = lane&15
  #pragma unroll
  for (int mt = 0; mt < MT; ++mt) {
    const int mOut = mBase0 + mt * 16 + (laneHi << 3);
    #pragma unroll
    for (int nt = 0; nt < NT; ++nt) {
      int ch = (nt << 4) + nLane;
      if (ch < Cout) {
        float bv = bias[ch];
        float* op = out + (size_t)ch * IHW + mOut;
        #pragma unroll
        for (int r = 0; r < 8; ++r) op[r] = acc[mt][nt][r] + bv;
      }
    }
  }
}

// ---------------------------------------------------------------------------
// 5. Whole-tensor mean / inv-std (block b -> stats[2b]=mean, stats[2b+1]=rstd)
// ---------------------------------------------------------------------------
__global__ __launch_bounds__(256) void reduce_stats_kernel(
    const float* __restrict__ buf, int rangeLen, float* __restrict__ stats) {
  __shared__ float ssum[256], ssq[256];
  const float* base = buf + (size_t)blockIdx.x * rangeLen;
  float s = 0.f, q = 0.f;
  for (int i = threadIdx.x; i < rangeLen; i += 256) {
    float v = base[i];
    s += v; q += v * v;
  }
  ssum[threadIdx.x] = s; ssq[threadIdx.x] = q;
  __syncthreads();
  for (int st = 128; st > 0; st >>= 1) {
    if (threadIdx.x < st) {
      ssum[threadIdx.x] += ssum[threadIdx.x + st];
      ssq[threadIdx.x]  += ssq[threadIdx.x + st];
    }
    __syncthreads();
  }
  if (threadIdx.x == 0) {
    float n = (float)rangeLen;
    float m = ssum[0] / n;
    float var = ssq[0] / n - m * m;
    stats[blockIdx.x * 2 + 0] = m;
    stats[blockIdx.x * 2 + 1] = rsqrtf(var + 1e-5f);
  }
}

// ---------------------------------------------------------------------------
// 6. r/u: rh = sigmoid(gn(r))*h -> f16 mirror; u -> f32 buf
// ---------------------------------------------------------------------------
__global__ __launch_bounds__(256) void normalize_ru_kernel(
    const float* __restrict__ f, const float* __restrict__ h,
    const float* __restrict__ rg, const float* __restrict__ rb,
    const float* __restrict__ ug, const float* __restrict__ ub,
    const float* __restrict__ stats,
    _Float16* __restrict__ rhDst, int rhStride, int rhOff,
    float* __restrict__ ubuf, int cout) {
  int total = cout * IHW;
  int i = blockIdx.x * 256 + threadIdx.x;
  if (i >= total) return;
  int c = i / IHW;
  int p = i - c * IHW;
  float rr = (f[i] - stats[0]) * stats[1] * rg[c] + rb[c];
  float r  = 1.0f / (1.0f + expf(-rr));
  rhDst[(size_t)p * rhStride + rhOff + c] = (_Float16)(r * h[i]);
  float uu = (f[total + i] - stats[2]) * stats[3] * ug[c] + ub[c];
  ubuf[i]  = 1.0f / (1.0f + expf(-uu));
}

// ---------------------------------------------------------------------------
// 7. Finalize: y=tanh(gn(o)); h = u*h+(1-u)*y; f16 mirrors (<=3 dests)
// ---------------------------------------------------------------------------
__global__ __launch_bounds__(256) void finalize_kernel(
    const float* __restrict__ o, const float* __restrict__ ubuf,
    const float* __restrict__ og, const float* __restrict__ obb,
    const float* __restrict__ stats, float* __restrict__ h, int cout,
    _Float16* __restrict__ m0, int s0, int o0,
    _Float16* __restrict__ m1, int s1, int o1,
    _Float16* __restrict__ m2, int s2, int o2) {
  int total = cout * IHW;
  int i = blockIdx.x * 256 + threadIdx.x;
  if (i >= total) return;
  int c = i / IHW;
  int p = i - c * IHW;
  float y = tanhf((o[i] - stats[0]) * stats[1] * og[c] + obb[c]);
  float u = ubuf[i];
  float hn = u * h[i] + (1.0f - u) * y;
  h[i] = hn;
  _Float16 hh = (_Float16)hn;
  m0[(size_t)p * s0 + o0 + c] = hh;
  if (m1) m1[(size_t)p * s1 + o1 + c] = hh;
  if (m2) m2[(size_t)p * s2 + o2 + c] = hh;
}

// ---------------------------------------------------------------------------
// 8. Final: per-pixel argmax over 96 regs, 1/sum(exp), depth lookup
// ---------------------------------------------------------------------------
__global__ __launch_bounds__(256) void softmax_depth_kernel(
    const float* __restrict__ regs, const float* __restrict__ depth_values,
    float* __restrict__ depth_out, float* __restrict__ prob_out) {
  int p = blockIdx.x * 256 + threadIdx.x;
  if (p >= IHW) return;
  float mx = -3.402823466e+38f;
  int am = 0;
  for (int d = 0; d < NDEPTH; ++d) {
    float v = regs[(size_t)d * IHW + p];
    if (v > mx) { mx = v; am = d; }
  }
  float se = 0.f;
  for (int d = 0; d < NDEPTH; ++d)
    se += expf(regs[(size_t)d * IHW + p] - mx);
  depth_out[p] = depth_values[am];
  prob_out[p]  = 1.0f / se;
}

// ---------------------------------------------------------------------------
// Host-side orchestration
// ---------------------------------------------------------------------------
static inline int grid256(int n) { return (n + 255) / 256; }

extern "C" void kernel_launch(void* const* d_in, const int* in_sizes, int n_in,
                              void* d_out, int out_size, void* d_ws, size_t ws_size,
                              hipStream_t stream) {
  (void)in_sizes; (void)n_in; (void)out_size; (void)ws_size;

  const float* features     = (const float*)d_in[0];   // (3,32,128,160)
  const float* Hs           = (const float*)d_in[1];   // (2,96,3,3)
  const float* depth_values = (const float*)d_in[2];   // (96)
  struct GruP {
    const float *gw, *gb, *rg, *rb, *ug, *ub, *ow, *ob, *og, *obb;
  };
  auto loadP = [&](int base) {
    GruP P;
    P.gw  = (const float*)d_in[base + 0]; P.gb  = (const float*)d_in[base + 1];
    P.rg  = (const float*)d_in[base + 2]; P.rb  = (const float*)d_in[base + 3];
    P.ug  = (const float*)d_in[base + 4]; P.ub  = (const float*)d_in[base + 5];
    P.ow  = (const float*)d_in[base + 6]; P.ob  = (const float*)d_in[base + 7];
    P.og  = (const float*)d_in[base + 8]; P.obb = (const float*)d_in[base + 9];
    return P;
  };
  GruP P1 = loadP(3), P2 = loadP(13), P3 = loadP(23);
  const float* pc_w = (const float*)d_in[33];
  const float* pc_b = (const float*)d_in[34];

  float* out       = (float*)d_out;             // [96*HW | HW depth | HW prob]
  float* regs      = out;
  float* depth_out = out + (size_t)NDEPTH * IHW;
  float* prob_out  = depth_out + IHW;

  // ---- workspace carve (256B-aligned bump) ----
  char* wsp = (char*)d_ws;
  auto alloc = [&](size_t bytes) -> char* {
    char* p = wsp;
    wsp += (bytes + 255) & ~(size_t)255;
    return p;
  };
  const size_t f16Elems = (size_t)IHW * (48 + 48 + 24 + 24 + 8 + 8 + 8);
  _Float16* f16base = (_Float16*)alloc(f16Elems * sizeof(_Float16));
  _Float16* xh1_g = f16base;                       // [IHW][48] x | h1
  _Float16* xh1_o = xh1_g + (size_t)IHW * 48;      // [IHW][48] x | r*h1
  _Float16* xh2_g = xh1_o + (size_t)IHW * 48;      // [IHW][24] h1 | h2 | pad
  _Float16* xh2_o = xh2_g + (size_t)IHW * 24;      // [IHW][24] h1 | r*h2 | pad
  _Float16* xh3_g = xh2_o + (size_t)IHW * 24;      // [IHW][8]  h2 | h3 | pad
  _Float16* xh3_o = xh3_g + (size_t)IHW * 8;       // [IHW][8]  h2 | r*h3 | pad
  _Float16* xpc   = xh3_o + (size_t)IHW * 8;       // [IHW][8]  h3 | pad

  float* Hw    = (float*)alloc(NDEPTH * 2 * 9 * sizeof(float));
  float* fbuf  = (float*)alloc((size_t)32 * IHW * sizeof(float));
  float* obuf  = (float*)alloc((size_t)16 * IHW * sizeof(float));
  float* ubuf  = (float*)alloc((size_t)16 * IHW * sizeof(float));
  float* h1    = (float*)alloc((size_t)(16 + 4 + 2) * IHW * sizeof(float));
  float* h2    = h1 + (size_t)16 * IHW;
  float* h3    = h2 + (size_t)4 * IHW;
  float* stats  = (float*)alloc(4 * sizeof(float));
  float* stats2 = (float*)alloc(2 * sizeof(float));
  _Float16* Wg1 = (_Float16*)alloc((size_t)32 * 448 * sizeof(_Float16));
  _Float16* Wo1 = (_Float16*)alloc((size_t)16 * 448 * sizeof(_Float16));
  _Float16* Wg2 = (_Float16*)alloc((size_t)16 * 224 * sizeof(_Float16));
  _Float16* Wo2 = (_Float16*)alloc((size_t)16 * 224 * sizeof(_Float16));
  _Float16* Wg3 = (_Float16*)alloc((size_t)16 * 96 * sizeof(_Float16));
  _Float16* Wo3 = (_Float16*)alloc((size_t)16 * 96 * sizeof(_Float16));
  _Float16* Wpc = (_Float16*)alloc((size_t)16 * 96 * sizeof(_Float16));

  const int convGrid = IHW / (16 * 2 * 4);  // MT=2, 4 waves/block -> 160

  // ---- precompute ----
  hipLaunchKernelGGL(compute_hw_kernel, dim3(1), dim3(192), 0, stream, Hs, Hw, NDEPTH);
  hipLaunchKernelGGL(zero_kernel, dim3(grid256((int)(f16Elems / 2))), dim3(256), 0,
                     stream, (float*)f16base, (int)(f16Elems / 2));
  hipLaunchKernelGGL(zero_kernel, dim3(grid256(22 * IHW)), dim3(256), 0, stream,
                     h1, 22 * IHW);
  hipLaunchKernelGGL(pack_w_kernel, dim3(grid256(32 * 448)), dim3(256), 0, stream, P1.gw, Wg1, 32, 48, 48, 448, 32);
  hipLaunchKernelGGL(pack_w_kernel, dim3(grid256(16 * 448)), dim3(256), 0, stream, P1.ow, Wo1, 16, 48, 48, 448, 16);
  hipLaunchKernelGGL(pack_w_kernel, dim3(grid256(16 * 224)), dim3(256), 0, stream, P2.gw, Wg2,  8, 20, 24, 224, 16);
  hipLaunchKernelGGL(pack_w_kernel, dim3(grid256(16 * 224)), dim3(256), 0, stream, P2.ow, Wo2,  4, 20, 24, 224, 16);
  hipLaunchKernelGGL(pack_w_kernel, dim3(grid256(16 *  96)), dim3(256), 0, stream, P3.gw, Wg3,  4,  6,  8,  96, 16);
  hipLaunchKernelGGL(pack_w_kernel, dim3(grid256(16 *  96)), dim3(256), 0, stream, P3.ow, Wo3,  2,  6,  8,  96, 16);
  hipLaunchKernelGGL(pack_w_kernel, dim3(grid256(16 *  96)), dim3(256), 0, stream, pc_w,  Wpc,  1,  2,  8,  96, 16);

  // ---- 96 sequential depth steps ----
  for (int d = 0; d < NDEPTH; ++d) {
    hipLaunchKernelGGL(warp_cost_kernel, dim3(grid256(IHW)), dim3(256), 0, stream,
                       features, Hw, d, xh1_g, xh1_o);

    // ---------------- GRU1: cin=32, co=16, CP=48 ----------------
    hipLaunchKernelGGL((conv3x3_wmma_kernel<2, 2, 48>), dim3(convGrid), dim3(128), 0, stream,
                       xh1_g, Wg1, P1.gb, fbuf, 32);
    hipLaunchKernelGGL(reduce_stats_kernel, dim3(2), dim3(256), 0, stream,
                       fbuf, 16 * IHW, stats);
    hipLaunchKernelGGL(normalize_ru_kernel, dim3(grid256(16 * IHW)), dim3(256), 0, stream,
                       fbuf, h1, P1.rg, P1.rb, P1.ug, P1.ub, stats,
                       xh1_o, 48, 32, ubuf, 16);
    hipLaunchKernelGGL((conv3x3_wmma_kernel<1, 2, 48>), dim3(convGrid), dim3(128), 0, stream,
                       xh1_o, Wo1, P1.ob, obuf, 16);
    hipLaunchKernelGGL(reduce_stats_kernel, dim3(1), dim3(256), 0, stream,
                       obuf, 16 * IHW, stats2);
    hipLaunchKernelGGL(finalize_kernel, dim3(grid256(16 * IHW)), dim3(256), 0, stream,
                       obuf, ubuf, P1.og, P1.obb, stats2, h1, 16,
                       xh1_g, 48, 32, xh2_g, 24, 0, xh2_o, 24, 0);

    // ---------------- GRU2: cin=16, co=4, CP=24 ----------------
    hipLaunchKernelGGL((conv3x3_wmma_kernel<1, 2, 24>), dim3(convGrid), dim3(128), 0, stream,
                       xh2_g, Wg2, P2.gb, fbuf, 8);
    hipLaunchKernelGGL(reduce_stats_kernel, dim3(2), dim3(256), 0, stream,
                       fbuf, 4 * IHW, stats);
    hipLaunchKernelGGL(normalize_ru_kernel, dim3(grid256(4 * IHW)), dim3(256), 0, stream,
                       fbuf, h2, P2.rg, P2.rb, P2.ug, P2.ub, stats,
                       xh2_o, 24, 16, ubuf, 4);
    hipLaunchKernelGGL((conv3x3_wmma_kernel<1, 2, 24>), dim3(convGrid), dim3(128), 0, stream,
                       xh2_o, Wo2, P2.ob, obuf, 4);
    hipLaunchKernelGGL(reduce_stats_kernel, dim3(1), dim3(256), 0, stream,
                       obuf, 4 * IHW, stats2);
    hipLaunchKernelGGL(finalize_kernel, dim3(grid256(4 * IHW)), dim3(256), 0, stream,
                       obuf, ubuf, P2.og, P2.obb, stats2, h2, 4,
                       xh2_g, 24, 16, xh3_g, 8, 0, xh3_o, 8, 0);

    // ---------------- GRU3: cin=4, co=2, CP=8 ----------------
    hipLaunchKernelGGL((conv3x3_wmma_kernel<1, 2, 8>), dim3(convGrid), dim3(128), 0, stream,
                       xh3_g, Wg3, P3.gb, fbuf, 4);
    hipLaunchKernelGGL(reduce_stats_kernel, dim3(2), dim3(256), 0, stream,
                       fbuf, 2 * IHW, stats);
    hipLaunchKernelGGL(normalize_ru_kernel, dim3(grid256(2 * IHW)), dim3(256), 0, stream,
                       fbuf, h3, P3.rg, P3.rb, P3.ug, P3.ub, stats,
                       xh3_o, 8, 4, ubuf, 2);
    hipLaunchKernelGGL((conv3x3_wmma_kernel<1, 2, 8>), dim3(convGrid), dim3(128), 0, stream,
                       xh3_o, Wo3, P3.ob, obuf, 2);
    hipLaunchKernelGGL(reduce_stats_kernel, dim3(1), dim3(256), 0, stream,
                       obuf, 2 * IHW, stats2);
    hipLaunchKernelGGL(finalize_kernel, dim3(grid256(2 * IHW)), dim3(256), 0, stream,
                       obuf, ubuf, P3.og, P3.obb, stats2, h3, 2,
                       xh3_g, 8, 4, xpc, 8, 0, (_Float16*)nullptr, 0, 0);

    // reg conv: 2 -> 1 channel, straight into prob_volume slice d
    hipLaunchKernelGGL((conv3x3_wmma_kernel<1, 2, 8>), dim3(convGrid), dim3(128), 0, stream,
                       xpc, Wpc, pc_b, regs + (size_t)d * IHW, 1);
  }

  // ---- softmax / argmax / depth lookup ----
  hipLaunchKernelGGL(softmax_depth_kernel, dim3(grid256(IHW)), dim3(256), 0, stream,
                     regs, depth_values, depth_out, prob_out);
}
